// ResidualMambaTokenStage_86878598463587
// MI455X (gfx1250) — matrix operand order repaired and checked by reading, not compiled, not run on hardware
//
#include <hip/hip_runtime.h>
#include <hip/hip_bf16.h>
#include <math.h>

// ---------------------------------------------------------------- constants
#define BB       4
#define CIN      3
#define HH       512
#define WW       512
#define DM       512
#define PP       16
#define NBLK     2
#define D_INNER  1024
#define D_STATE  16
#define D_CONV   4
#define DT_RANK  32
#define LTOK     1024            // (H/P)*(W/P) = 32*32
#define NTOK     (BB * LTOK)     // 4096
#define KC       (CIN * PP * PP) // 768

// WMMA tile config: block = 128 threads = 4 waves.
// Block tile: M=32, N=64, K-step 32. Each wave owns a 16-wide N slice and
// issues two WMMAs per K-step (M halves) sharing one B fragment.
#define TM 32
#define TN 64
#define KS 32
#define KPAD 36   // even -> b32-packed LDS stores; 72B row stride breaks bank alignment

typedef __attribute__((ext_vector_type(16))) __bf16 v16bf;
typedef __attribute__((ext_vector_type(8)))  float  v8f;

// pack two f32 -> packed bf16 dword (v_cvt_pk_bf16_f32)
__device__ __forceinline__ unsigned pack_bf16(float a, float b) {
  union { __bf16 h[2]; unsigned u; } r;
  r.h[0] = (__bf16)a; r.h[1] = (__bf16)b;
  return r.u;
}

// ---------------------------------------------------------------- fragments
// A fragment (16x32 bf16, MxK): lane l holds row m=l%16.
// VGPR p (0..3): K = (l/16)*8 + {2p,2p+1};  VGPR 4+p: K = 16 + (l/16)*8 + {2p,2p+1}
__device__ __forceinline__ v16bf load_frag_a(const __bf16* As, int lane) {
  v16bf a;
  const int m    = lane & 15;
  const int koff = (lane >> 4) * 8;
  const __bf16* row = As + m * KPAD;
#pragma unroll
  for (int p = 0; p < 4; ++p) {
    a[2 * p]         = row[koff + 2 * p];
    a[2 * p + 1]     = row[koff + 2 * p + 1];
    a[8 + 2 * p]     = row[16 + koff + 2 * p];
    a[8 + 2 * p + 1] = row[16 + koff + 2 * p + 1];
  }
  return a;
}

// B fragment (32x16 bf16, KxN): lane l holds column n=l%16; K = (l/16)*8 + j, 16 + (l/16)*8 + (j-8).
// Bs laid out [n][k], stride KPAD.
__device__ __forceinline__ v16bf load_frag_b(const __bf16* Bs, int lane) {
  v16bf b;
  const int n    = lane & 15;
  const int koff = (lane >> 4) * 8;
  const __bf16* row = Bs + n * KPAD + koff;
#pragma unroll
  for (int j = 0; j < 16; ++j) b[j] = row[(j < 8) ? j : (8 + j)];
  return b;
}

__device__ __forceinline__ float softplus_f(float x) {
  return (x > 20.f) ? x : __logf(1.f + __expf(x));
}
__device__ __forceinline__ float silu_f(float x) {
  return x / (1.f + __expf(-x));
}

// ---------------------------------------------------------------- generic GEMM
// C[M,N] (op) A[M,K] * W[N,K]^T  (A row-major stride lda, W row-major stride ldw)
// epi: 0 = store, 1 = store softplus(v + bias[n]), 2 = residual accumulate C += v
__global__ __launch_bounds__(128) void k_gemm(const float* __restrict__ A, int lda,
                                              const float* __restrict__ W, int ldw,
                                              float* __restrict__ C, int ldc,
                                              int K,
                                              const float* __restrict__ bias, int epi) {
  __shared__ __bf16 As[TM * KPAD];
  __shared__ __bf16 Bs[TN * KPAD];

  const int tid  = threadIdx.x;
  const int lane = tid & 31;
  const int wave = tid >> 5;
  const int m0   = blockIdx.x * TM;
  const int n0   = blockIdx.y * TN;

  v8f acc0 = {}, acc1 = {};

  for (int k0 = 0; k0 < K; k0 += KS) {
    // stage A tile: 32x32 f32 -> packed bf16; 512 pairs, 4 per thread (b64 loads)
#pragma unroll
    for (int i = 0; i < 4; ++i) {
      const int idx = i * 128 + tid;
      const int r = idx >> 4, kp = idx & 15;
      const float2 v = *(const float2*)&A[(size_t)(m0 + r) * lda + k0 + kp * 2];
      *(unsigned*)&As[r * KPAD + kp * 2] = pack_bf16(v.x, v.y);
    }
    // stage B tile: 64x32 f32 -> packed bf16; 1024 pairs, 8 per thread
#pragma unroll
    for (int i = 0; i < 8; ++i) {
      const int idx = i * 128 + tid;
      const int r = idx >> 4, kp = idx & 15;
      const float2 v = *(const float2*)&W[(size_t)(n0 + r) * ldw + k0 + kp * 2];
      *(unsigned*)&Bs[r * KPAD + kp * 2] = pack_bf16(v.x, v.y);
    }
    if (k0 + KS < K) {  // hint next K-slab (global_prefetch_b8)
      __builtin_prefetch(&A[(size_t)(m0 + (tid & 31)) * lda + k0 + KS], 0, 1);
      __builtin_prefetch(&W[(size_t)(n0 + (tid >> 1)) * ldw + k0 + KS], 0, 1);
    }
    __syncthreads();
    v16bf a0 = load_frag_a(As, lane);
    v16bf a1 = load_frag_a(As + 16 * KPAD, lane);
    v16bf bf = load_frag_b(Bs + wave * 16 * KPAD, lane);
    acc0 = __builtin_amdgcn_wmma_f32_16x16x32_bf16(false, a0, false, bf,
                                                   (short)0, acc0, false, false);
    acc1 = __builtin_amdgcn_wmma_f32_16x16x32_bf16(false, a1, false, bf,
                                                   (short)0, acc1, false, false);
    __syncthreads();
  }

  // epilogue: lane l holds n = l%16, rows m = (l/16)*8 + r per 16x16 tile
  const int n  = n0 + wave * 16 + (lane & 15);
  const int mh = (lane >> 4) * 8;
  float bn = (epi == 1) ? bias[n] : 0.f;
#pragma unroll
  for (int r = 0; r < 8; ++r) {
    const int mA = m0 + mh + r;
    const int mB = m0 + 16 + mh + r;
    float v0 = acc0[r], v1 = acc1[r];
    if (epi == 1) { v0 = softplus_f(v0 + bn); v1 = softplus_f(v1 + bn); }
    float* d0 = &C[(size_t)mA * ldc + n];
    float* d1 = &C[(size_t)mB * ldc + n];
    if (epi == 2) { *d0 += v0; *d1 += v1; } else { *d0 = v0; *d1 = v1; }
  }
}

// ---------------------------------------------------------------- patch-embed GEMM
// tokens_raw[t, d] = sum_k im2col(x)[t,k] * conv_w[d,k] + conv_b[d]
__global__ __launch_bounds__(128) void k_patch(const float* __restrict__ x,
                                               const float* __restrict__ cw,
                                               const float* __restrict__ cb,
                                               float* __restrict__ out) {
  __shared__ __bf16 As[TM * KPAD];
  __shared__ __bf16 Bs[TN * KPAD];

  const int tid  = threadIdx.x;
  const int lane = tid & 31;
  const int wave = tid >> 5;
  const int m0   = blockIdx.x * TM;
  const int n0   = blockIdx.y * TN;

  v8f acc0 = {}, acc1 = {};

  for (int k0 = 0; k0 < KC; k0 += KS) {
    // im2col gather of A tile (pairs stay within one patch row: k even, k+1 same (ci,pr))
#pragma unroll
    for (int i = 0; i < 4; ++i) {
      const int idx = i * 128 + tid;
      const int r = idx >> 4, kp = idx & 15;
      const int t  = m0 + r;
      const int b  = t >> 10;
      const int l  = t & 1023;
      const int oh = l >> 5, ow = l & 31;
      const int k  = k0 + kp * 2;
      const int ci = k >> 8;
      const int pr = (k >> 4) & 15, pc = k & 15;
      const float2 v = *(const float2*)&x[(((size_t)(b * CIN + ci) * HH) +
                                           (oh * PP + pr)) * WW + (ow * PP + pc)];
      *(unsigned*)&As[r * KPAD + kp * 2] = pack_bf16(v.x, v.y);
    }
#pragma unroll
    for (int i = 0; i < 8; ++i) {
      const int idx = i * 128 + tid;
      const int r = idx >> 4, kp = idx & 15;
      const float2 v = *(const float2*)&cw[(size_t)(n0 + r) * KC + k0 + kp * 2];
      *(unsigned*)&Bs[r * KPAD + kp * 2] = pack_bf16(v.x, v.y);
    }
    __syncthreads();
    v16bf a0 = load_frag_a(As, lane);
    v16bf a1 = load_frag_a(As + 16 * KPAD, lane);
    v16bf bf = load_frag_b(Bs + wave * 16 * KPAD, lane);
    acc0 = __builtin_amdgcn_wmma_f32_16x16x32_bf16(false, a0, false, bf,
                                                   (short)0, acc0, false, false);
    acc1 = __builtin_amdgcn_wmma_f32_16x16x32_bf16(false, a1, false, bf,
                                                   (short)0, acc1, false, false);
    __syncthreads();
  }

  const int n  = n0 + wave * 16 + (lane & 15);
  const int mh = (lane >> 4) * 8;
  const float bn = cb[n];
#pragma unroll
  for (int r = 0; r < 8; ++r) {
    out[(size_t)(m0 + mh + r) * DM + n]      = acc0[r] + bn;
    out[(size_t)(m0 + 16 + mh + r) * DM + n] = acc1[r] + bn;
  }
}

// ---------------------------------------------------------------- row LayerNorm (rowlen = 512)
__global__ __launch_bounds__(128) void k_row_ln(const float* __restrict__ in,
                                                float* __restrict__ out,
                                                const float* __restrict__ g,
                                                const float* __restrict__ b) {
  __shared__ float rs[128], rq[128];
  const int tid = threadIdx.x;
  const float* row = in + (size_t)blockIdx.x * DM;
  float vals[4], s = 0.f, q = 0.f;
#pragma unroll
  for (int i = 0; i < 4; ++i) {
    const float v = row[tid + i * 128];
    vals[i] = v; s += v; q += v * v;
  }
  rs[tid] = s; rq[tid] = q;
  __syncthreads();
  for (int off = 64; off > 0; off >>= 1) {
    if (tid < off) { rs[tid] += rs[tid + off]; rq[tid] += rq[tid + off]; }
    __syncthreads();
  }
  const float mu  = rs[0] * (1.f / DM);
  const float var = rq[0] * (1.f / DM) - mu * mu;
  const float inv = rsqrtf(var + 1e-5f);
  float* orow = out + (size_t)blockIdx.x * DM;
#pragma unroll
  for (int i = 0; i < 4; ++i) {
    const int c = tid + i * 128;
    orow[c] = (vals[i] - mu) * inv * g[c] + b[c];
  }
}

// ---------------------------------------------------------------- causal depthwise conv1d + SiLU
// u[t,d] = silu( sum_j xin[t-3+j, d] * cw[d, j] + cb[d] ),  xin = xz[:, 0:1024] (row stride 2048)
__global__ __launch_bounds__(256) void k_conv_silu(const float* __restrict__ xz,
                                                   const float* __restrict__ cw,
                                                   const float* __restrict__ cb,
                                                   float* __restrict__ u) {
  const int idx = blockIdx.x * 256 + threadIdx.x;   // over NTOK * D_INNER
  const int d = idx & (D_INNER - 1);
  const int t = idx >> 10;
  const int l = t & (LTOK - 1);
  float acc = cb[d];
#pragma unroll
  for (int j = 0; j < D_CONV; ++j) {
    const int ls = l - (D_CONV - 1) + j;
    if (ls >= 0)
      acc += xz[(size_t)(t - (D_CONV - 1) + j) * (2 * D_INNER) + d] * cw[d * D_CONV + j];
  }
  u[(size_t)idx] = silu_f(acc);
}

// ---------------------------------------------------------------- selective scan + skip + gate
// per (b,d): h[s] = exp(delta*A[s])*h[s] + delta*B[l,s]*u;  y_l = h.C_l ; y = (y + u*Dp)*silu(z)
// Double-buffered B/C stage: one barrier per timestep, next-step load overlaps math.
__global__ __launch_bounds__(256) void k_scan(const float* __restrict__ delta,
                                              const float* __restrict__ u,
                                              const float* __restrict__ xdbl,
                                              const float* __restrict__ xz,
                                              const float* __restrict__ alog,
                                              const float* __restrict__ Dp,
                                              float* __restrict__ y) {
  const int b = blockIdx.y;
  const int d = blockIdx.x * 256 + threadIdx.x;
  __shared__ float sBC[2][2 * D_STATE];

  float Ad[D_STATE], h[D_STATE];
#pragma unroll
  for (int s = 0; s < D_STATE; ++s) {
    Ad[s] = -__expf(alog[d * D_STATE + s]);
    h[s]  = 0.f;
  }
  const float dp = Dp[d];
  const int t0 = b * LTOK;

  if (threadIdx.x < 2 * D_STATE)
    sBC[0][threadIdx.x] = xdbl[(size_t)t0 * 64 + DT_RANK + threadIdx.x];
  __syncthreads();

  for (int l = 0; l < LTOK; ++l) {
    const int t   = t0 + l;
    const int cur = l & 1;
    if ((l + 1 < LTOK) && threadIdx.x < 2 * D_STATE)
      sBC[cur ^ 1][threadIdx.x] = xdbl[(size_t)(t + 1) * 64 + DT_RANK + threadIdx.x];

    const float dlt = delta[(size_t)t * D_INNER + d];
    const float uu  = u[(size_t)t * D_INNER + d];
    float acc = 0.f;
#pragma unroll
    for (int s = 0; s < D_STATE; ++s) {
      h[s] = __expf(dlt * Ad[s]) * h[s] + dlt * sBC[cur][s] * uu;
      acc += h[s] * sBC[cur][D_STATE + s];
    }
    float yy = acc + uu * dp;
    const float zz = xz[(size_t)t * (2 * D_INNER) + D_INNER + d];
    yy *= silu_f(zz);
    y[(size_t)t * D_INNER + d] = yy;
    __syncthreads();   // next buffer ready / current reads done
  }
}

// ---------------------------------------------------------------- launch
extern "C" void kernel_launch(void* const* d_in, const int* in_sizes, int n_in,
                              void* d_out, int out_size, void* d_ws, size_t ws_size,
                              hipStream_t stream) {
  const float* x         = (const float*)d_in[0];
  const float* conv_w    = (const float*)d_in[1];
  const float* conv_b    = (const float*)d_in[2];
  const float* pe_g      = (const float*)d_in[3];
  const float* pe_b      = (const float*)d_in[4];
  const float* ln_g      = (const float*)d_in[5];
  const float* ln_b      = (const float*)d_in[6];
  const float* in_proj_w = (const float*)d_in[7];
  const float* conv1d_w  = (const float*)d_in[8];
  const float* conv1d_b  = (const float*)d_in[9];
  const float* x_proj_w  = (const float*)d_in[10];
  const float* dt_proj_w = (const float*)d_in[11];
  const float* dt_proj_b = (const float*)d_in[12];
  const float* A_log     = (const float*)d_in[13];
  const float* Dp        = (const float*)d_in[14];
  const float* out_proj_w= (const float*)d_in[15];

  float* tok = (float*)d_out;                       // running tokens (NTOK x DM)
  float* ws  = (float*)d_ws;
  float* t_ln  = ws;                                // NTOK*DM
  float* xz    = t_ln  + (size_t)NTOK * DM;         // NTOK*2*D_INNER
  float* u     = xz    + (size_t)NTOK * 2 * D_INNER;// NTOK*D_INNER
  float* xdbl  = u     + (size_t)NTOK * D_INNER;    // NTOK*64
  float* delta = xdbl  + (size_t)NTOK * 64;         // NTOK*D_INNER
  float* yb    = delta + (size_t)NTOK * D_INNER;    // NTOK*D_INNER

  const dim3 blk(128);

  // patch-embed GEMM + bias, then channel LN (per token row over DM)
  k_patch<<<dim3(NTOK / TM, DM / TN), blk, 0, stream>>>(x, conv_w, conv_b, tok);
  k_row_ln<<<NTOK, 128, 0, stream>>>(tok, tok, pe_g, pe_b);

  for (int i = 0; i < NBLK; ++i) {
    // pre-LN
    k_row_ln<<<NTOK, 128, 0, stream>>>(tok, t_ln, ln_g + i * DM, ln_b + i * DM);
    // in_proj: (4096x512)x(2048x512)^T -> xz
    k_gemm<<<dim3(NTOK / TM, (2 * D_INNER) / TN), blk, 0, stream>>>(
        t_ln, DM, in_proj_w + (size_t)i * 2 * D_INNER * DM, DM,
        xz, 2 * D_INNER, DM, nullptr, 0);
    // causal depthwise conv + SiLU -> u
    k_conv_silu<<<(NTOK * D_INNER) / 256, 256, 0, stream>>>(
        xz, conv1d_w + (size_t)i * D_INNER * D_CONV, conv1d_b + (size_t)i * D_INNER, u);
    // x_proj: (4096x1024)x(64x1024)^T -> xdbl
    k_gemm<<<dim3(NTOK / TM, 64 / TN), blk, 0, stream>>>(
        u, D_INNER, x_proj_w + (size_t)i * 64 * D_INNER, D_INNER,
        xdbl, 64, D_INNER, nullptr, 0);
    // dt_proj + bias + softplus: (4096x32)x(1024x32)^T -> delta
    k_gemm<<<dim3(NTOK / TM, D_INNER / TN), blk, 0, stream>>>(
        xdbl, 64, dt_proj_w + (size_t)i * D_INNER * DT_RANK, DT_RANK,
        delta, D_INNER, DT_RANK, dt_proj_b + (size_t)i * D_INNER, 1);
    // selective scan + skip + gating -> yb
    k_scan<<<dim3(D_INNER / 256, BB), 256, 0, stream>>>(
        delta, u, xdbl, xz, A_log + (size_t)i * D_INNER * D_STATE,
        Dp + (size_t)i * D_INNER, yb);
    // out_proj + residual accumulate into tokens
    k_gemm<<<dim3(NTOK / TM, DM / TN), blk, 0, stream>>>(
        yb, D_INNER, out_proj_w + (size_t)i * DM * D_INNER, D_INNER,
        tok, DM, D_INNER, nullptr, 2);
  }
}
